// MultiHeadGlobalAttention_3427383902580
// MI455X (gfx1250) — compile-verified, matrix operand order checked by reading, not run on hardware
//
#include <hip/hip_runtime.h>
#include <hip/hip_bf16.h>
#include <float.h>

#define N_IN     64
#define HEADS    4
#define NSEG     1024
#define GATE_HID 256   // HEADS * N_IN
#define FEAT_DIM 128   // OUT_CH * HEADS
#define WAVES    8

typedef __attribute__((ext_vector_type(16))) _Float16 v16h;
typedef __attribute__((ext_vector_type(8)))  _Float16 v8h;
typedef __attribute__((ext_vector_type(8)))  float    v8f;

// K-offset inside a 32-wide K slab for packed-f16 element e of a lane in half h
// (ISA 7.12.2: lanes 0-15 hold K=0..7,16..23; lanes 16-31 hold K=8..15,24..31)
static __device__ __forceinline__ int koff(int e, int half) {
  return (e & 7) + ((e >> 3) << 4) + (half << 3);
}

static __device__ __forceinline__ v16h loadTile_f32(const float* rowPtr, int kBase, int half) {
  v16h a;
  #pragma unroll
  for (int e = 0; e < 16; ++e) a[e] = (_Float16)rowPtr[kBase + koff(e, half)];
  return a;
}

static __device__ __forceinline__ v16h loadTile_h(const _Float16* rowPtr, int kBase, int half) {
  v16h a;
  #pragma unroll
  for (int e = 0; e < 16; ++e) a[e] = rowPtr[kBase + koff(e, half)];
  return a;
}

static __device__ __forceinline__ v8f wmma_f16(v16h a, v16h b, v8f c) {
  return __builtin_amdgcn_wmma_f32_16x16x32_f16(false, a, false, b, (short)0, c, false, false);
}

// float atomic max via sign-split int/uint trick (no CAS loop)
static __device__ __forceinline__ void atomicMaxF(float* addr, float v) {
  if (v >= 0.0f) atomicMax((int*)addr, __float_as_int(v));
  else           atomicMin((unsigned int*)addr, (unsigned int)__float_as_int(v));
}

// CDNA5 async global->LDS copy (ASYNCcnt path); src/dst 16B aligned, bytes % 16 == 0
static __device__ __forceinline__ void asyncCopyToLds(_Float16* dst, const _Float16* src,
                                                      int nHalves, int tid, int nThreads) {
  const unsigned ldsBase = (unsigned)(uintptr_t)dst;   // addr[31:0] = LDS byte offset
  const int bytes = nHalves * 2;
  for (int off = tid * 16; off < bytes; off += nThreads * 16) {
    unsigned lds = ldsBase + off;
    const char* g = (const char*)src + off;
    asm volatile("global_load_async_to_lds_b128 %0, %1, off"
                 :: "v"(lds), "v"(g) : "memory");
  }
}

__global__ void prep_kernel(const float* __restrict__ gw1, const float* __restrict__ gw2,
                            const float* __restrict__ mw1, const float* __restrict__ mw2,
                            _Float16* gw1h, _Float16* gw2h, _Float16* mw1h, _Float16* mw2h) {
  int i = blockIdx.x * blockDim.x + threadIdx.x;
  if (i < GATE_HID * N_IN)    gw1h[i] = (_Float16)gw1[i];
  if (i < HEADS * GATE_HID)   gw2h[i] = (_Float16)gw2[i];
  if (i < FEAT_DIM * N_IN)    mw1h[i] = (_Float16)mw1[i];
  if (i < FEAT_DIM * FEAT_DIM) mw2h[i] = (_Float16)mw2[i];
}

__global__ void init_kernel(float* gmax, float* denom, float* out) {
  int i = blockIdx.x * blockDim.x + threadIdx.x;
  if (i < NSEG * HEADS) { gmax[i] = -FLT_MAX; denom[i] = 0.f; }
  if (i < NSEG * FEAT_DIM) out[i] = 0.f;
}

__global__ __launch_bounds__(WAVES * 32)
void gate_kernel(const float* __restrict__ x, const int* __restrict__ batch,
                 const _Float16* __restrict__ gw1h, const float* __restrict__ prelu_a,
                 const _Float16* __restrict__ gw2h,
                 float* __restrict__ gate, float* __restrict__ gmax,
                 _Float16* __restrict__ xh,          // optional f16 copy of x (may be null)
                 int n, int nTiles)
{
  __shared__ __attribute__((aligned(16))) _Float16 sW1[GATE_HID * N_IN];   // 32 KB
  __shared__ __attribute__((aligned(16))) _Float16 sW2[HEADS * GATE_HID];  //  2 KB
  asyncCopyToLds(sW1, gw1h, GATE_HID * N_IN, threadIdx.x, blockDim.x);
  asyncCopyToLds(sW2, gw2h, HEADS * GATE_HID, threadIdx.x, blockDim.x);
  asm volatile("s_wait_asynccnt 0" ::: "memory");
  __syncthreads();

  const float slope = prelu_a[0];
  const int lane = threadIdx.x & 31;
  const int wave = threadIdx.x >> 5;
  const int tile = blockIdx.x * WAVES + wave;
  if (tile >= nTiles) return;

  const int rowBase = tile * 16;
  const int half = lane >> 4;
  int myRow = rowBase + (lane & 15);
  if (myRow >= n) myRow = n - 1;                   // tail clamp (loads only)
  const float* xr = x + (size_t)myRow * N_IN;
  __builtin_prefetch(xr + (size_t)WAVES * 16 * N_IN, 0, 0);

  const v16h a0 = loadTile_f32(xr, 0, half);
  const v16h a1 = loadTile_f32(xr, 32, half);

  // emit the f16 copy of x for the second pass (two halves cover all 64 K)
  if (xh) {
    v8h* dst = (v8h*)(xh + (size_t)myRow * N_IN + half * 8);
    dst[0] = a0.lo;   // K = half*8 + 0..7
    dst[2] = a0.hi;   // K = 16 + half*8 ..
    dst[4] = a1.lo;   // K = 32 + half*8 ..
    dst[6] = a1.hi;   // K = 48 + half*8 ..
  }

  float gacc[8][HEADS];
  #pragma unroll
  for (int r = 0; r < 8; ++r)
    #pragma unroll
    for (int j = 0; j < HEADS; ++j) gacc[r][j] = 0.f;

  #pragma unroll
  for (int t = 0; t < GATE_HID / 16; ++t) {        // 16 N-tiles of the hidden layer
    const int col = t * 16 + (lane & 15);
    const _Float16* wr = sW1 + col * N_IN;
    v8f c = {};
    c = wmma_f16(a0, loadTile_h(wr, 0, half), c);
    c = wmma_f16(a1, loadTile_h(wr, 32, half), c);
    // PReLU, then fold the tiny [256 -> 4] GEMM into per-lane dot products
    #pragma unroll
    for (int r = 0; r < 8; ++r) {
      float h = c[r];
      h = (h >= 0.f) ? h : slope * h;
      #pragma unroll
      for (int j = 0; j < HEADS; ++j)
        gacc[r][j] += h * (float)sW2[j * GATE_HID + col];
    }
  }

  // sum over the 16 N-lanes within each half-wave
  #pragma unroll
  for (int m = 1; m < 16; m <<= 1)
    #pragma unroll
    for (int r = 0; r < 8; ++r)
      #pragma unroll
      for (int j = 0; j < HEADS; ++j)
        gacc[r][j] += __shfl_xor(gacc[r][j], m, 32);

  if ((lane & 15) == 0) {
    #pragma unroll
    for (int r = 0; r < 8; ++r) {
      const int row = rowBase + r + 8 * half;
      if (row >= n) continue;
      const int seg = batch[row];
      #pragma unroll
      for (int j = 0; j < HEADS; ++j) {
        const float g = gacc[r][j];
        gate[row * HEADS + j] = g;
        atomicMaxF(&gmax[seg * HEADS + j], g);
      }
    }
  }
}

__global__ __launch_bounds__(WAVES * 32)
void feat_kernel(const float* __restrict__ x, const _Float16* __restrict__ xh,
                 const int* __restrict__ batch,
                 const _Float16* __restrict__ mw1h, const float* __restrict__ mb1,
                 const _Float16* __restrict__ mw2h, const float* __restrict__ mb2,
                 const float* __restrict__ gate, const float* __restrict__ gmax,
                 float* __restrict__ denom, float* __restrict__ out,
                 int n, int nTiles)
{
  __shared__ __attribute__((aligned(16))) _Float16 sW1[FEAT_DIM * N_IN];      // 16 KB
  __shared__ float sB1[FEAT_DIM], sB2[FEAT_DIM];
  __shared__ __attribute__((aligned(16))) _Float16 sT1[WAVES][16 * FEAT_DIM]; // 32 KB
  asyncCopyToLds(sW1, mw1h, FEAT_DIM * N_IN, threadIdx.x, blockDim.x);
  for (int i = threadIdx.x; i < FEAT_DIM; i += blockDim.x) { sB1[i] = mb1[i]; sB2[i] = mb2[i]; }
  asm volatile("s_wait_asynccnt 0" ::: "memory");
  __syncthreads();

  const int lane = threadIdx.x & 31;
  const int wave = threadIdx.x >> 5;
  const int tile = blockIdx.x * WAVES + wave;
  if (tile >= nTiles) return;

  const int rowBase = tile * 16;
  const int half = lane >> 4;

  // softmax weights for the 8 C-rows this lane owns (esc==0 kills tail rows)
  int   segr[8];
  float esc[8][HEADS];
  #pragma unroll
  for (int r = 0; r < 8; ++r) {
    const int row = rowBase + r + 8 * half;
    const bool ok = row < n;
    const int rc = ok ? row : n - 1;
    const int seg = batch[rc];
    segr[r] = seg;
    #pragma unroll
    for (int h = 0; h < HEADS; ++h)
      esc[r][h] = ok ? __expf(gate[rc * HEADS + h] - gmax[seg * HEADS + h]) : 0.f;
  }
  if ((lane & 15) == 0) {                          // denom atomics deduped per half-wave
    #pragma unroll
    for (int r = 0; r < 8; ++r)
      #pragma unroll
      for (int h = 0; h < HEADS; ++h)
        atomicAdd(&denom[segr[r] * HEADS + h], esc[r][h]);
  }

  int myRow = rowBase + (lane & 15);
  if (myRow >= n) myRow = n - 1;
  v16h a0, a1;
  if (xh) {                                        // f16 copy written by gate pass
    const _Float16* xr = xh + (size_t)myRow * N_IN;
    __builtin_prefetch(xr + (size_t)WAVES * 16 * N_IN, 0, 0);
    a0 = loadTile_h(xr, 0, half);
    a1 = loadTile_h(xr, 32, half);
  } else {
    const float* xr = x + (size_t)myRow * N_IN;
    __builtin_prefetch(xr + (size_t)WAVES * 16 * N_IN, 0, 0);
    a0 = loadTile_f32(xr, 0, half);
    a1 = loadTile_f32(xr, 32, half);
  }

  // GEMM1: t1 = relu(x @ w1^T + b1), staged to LDS as f16 in row-major
  _Float16* t1 = &sT1[wave][0];
  #pragma unroll
  for (int t = 0; t < FEAT_DIM / 16; ++t) {
    const int col = t * 16 + (lane & 15);
    const _Float16* wr = sW1 + col * N_IN;
    v8f c = {};
    c = wmma_f16(a0, loadTile_h(wr, 0, half), c);
    c = wmma_f16(a1, loadTile_h(wr, 32, half), c);
    const float b = sB1[col];
    #pragma unroll
    for (int r = 0; r < 8; ++r) {
      float v = c[r] + b;
      v = (v > 0.f) ? v : 0.f;
      t1[(r + 8 * half) * FEAT_DIM + col] = (_Float16)v;
    }
  }
  // only this wave touches its sT1 slice: close the within-wave LDS RAW
  asm volatile("s_wait_dscnt 0" ::: "memory");

  // GEMM2: feat = t1 @ w2^T + b2  (K = 128 -> 4 k32 steps); B straight from f16 in L2
  v16h ta[4];
  const _Float16* t1row = t1 + (lane & 15) * FEAT_DIM;
  #pragma unroll
  for (int kk = 0; kk < 4; ++kk) ta[kk] = loadTile_h(t1row, kk * 32, half);

  #pragma unroll
  for (int t = 0; t < FEAT_DIM / 16; ++t) {
    const int col = t * 16 + (lane & 15);
    const _Float16* wr = mw2h + (size_t)col * FEAT_DIM;
    v8f c = {};
    #pragma unroll
    for (int kk = 0; kk < 4; ++kk)
      c = wmma_f16(ta[kk], loadTile_h(wr, kk * 32, half), c);
    const int head = col >> 5;
    const float b = sB2[col];
    #pragma unroll
    for (int r = 0; r < 8; ++r) {
      const float f = c[r] + b;
      atomicAdd(&out[(size_t)segr[r] * FEAT_DIM + col], esc[r][head] * f);
    }
  }
}

__global__ void finalize_kernel(float* out, const float* __restrict__ denom) {
  int i = blockIdx.x * blockDim.x + threadIdx.x;
  if (i >= NSEG * FEAT_DIM) return;
  const int seg  = i / FEAT_DIM;
  const int head = (i % FEAT_DIM) >> 5;
  out[i] = out[i] / (denom[seg * HEADS + head] + 1e-16f);
}

extern "C" void kernel_launch(void* const* d_in, const int* in_sizes, int n_in,
                              void* d_out, int out_size, void* d_ws, size_t ws_size,
                              hipStream_t stream) {
  const float* x    = (const float*)d_in[0];
  const int*   bat  = (const int*)  d_in[1];
  // d_in[2] = num_segments scalar (fixed at NSEG)
  const float* gw1  = (const float*)d_in[3];
  const float* pa   = (const float*)d_in[4];
  const float* gw2  = (const float*)d_in[5];
  const float* mw1  = (const float*)d_in[6];
  const float* mb1  = (const float*)d_in[7];
  const float* mw2  = (const float*)d_in[8];
  const float* mb2  = (const float*)d_in[9];
  float* out = (float*)d_out;

  const int n      = in_sizes[0] / N_IN;
  const int nTiles = (n + 15) / 16;

  // ws layout (all 16B aligned): gate | gmax | denom | f16 weights | optional f16 x
  float* gate  = (float*)d_ws;                           // nTiles*16*HEADS floats
  float* gmax  = gate + (size_t)nTiles * 16 * HEADS;     // NSEG*HEADS
  float* denom = gmax + NSEG * HEADS;                    // NSEG*HEADS
  _Float16* gw1h = (_Float16*)(denom + NSEG * HEADS);
  _Float16* gw2h = gw1h + GATE_HID * N_IN;
  _Float16* mw1h = gw2h + HEADS * GATE_HID;
  _Float16* mw2h = mw1h + FEAT_DIM * N_IN;
  _Float16* xh   = mw2h + FEAT_DIM * FEAT_DIM;
  const size_t needXh = (size_t)((char*)(xh + (size_t)n * N_IN) - (char*)d_ws);
  _Float16* xhUse = (ws_size >= needXh) ? xh : (_Float16*)nullptr;

  prep_kernel<<<(GATE_HID * N_IN + 255) / 256, 256, 0, stream>>>(gw1, gw2, mw1, mw2,
                                                                 gw1h, gw2h, mw1h, mw2h);
  init_kernel<<<(NSEG * FEAT_DIM + 255) / 256, 256, 0, stream>>>(gmax, denom, out);

  const int blocks = (nTiles + WAVES - 1) / WAVES;
  gate_kernel<<<blocks, WAVES * 32, 0, stream>>>(x, bat, gw1h, pa, gw2h,
                                                 gate, gmax, xhUse, n, nTiles);
  feat_kernel<<<blocks, WAVES * 32, 0, stream>>>(x, xhUse, bat, mw1h, mb1, mw2h, mb2,
                                                 gate, gmax, denom, out, n, nTiles);
  finalize_kernel<<<(NSEG * FEAT_DIM + 255) / 256, 256, 0, stream>>>(out, denom);
}
